// LangGCN_65309272703508
// MI455X (gfx1250) — matrix operand order repaired and checked by reading, not compiled, not run on hardware
//
#include <hip/hip_runtime.h>
#include <hip/hip_bf16.h>

#define N_NODES 100000
#define N_EDGES 3300000
#define IN_DIM 512
#define HID 512
#define OUT_DIM 256
#define BATCH 16

typedef __attribute__((ext_vector_type(16))) __bf16 v16bf;
typedef __attribute__((ext_vector_type(8)))  float  v8f;
typedef __attribute__((ext_vector_type(4)))  float  v4f;

// ---------------------------------------------------------------- zero ints
__global__ void zero_int(int* p, int n) {
    for (int i = blockIdx.x * blockDim.x + threadIdx.x; i < n;
         i += gridDim.x * blockDim.x) p[i] = 0;
}

// ------------------------------------------- W1 (f32 [k][n]) -> bf16 [n][k]
__global__ void conv_w1t(const float* __restrict__ W1, __bf16* __restrict__ W1t) {
    int idx = blockIdx.x * blockDim.x + threadIdx.x;   // 512*512
    int k = idx >> 9;
    int n = idx & 511;
    W1t[(size_t)n * IN_DIM + k] = (__bf16)W1[idx];
}

// ------------------------------------------------ CSR build: row histogram
__global__ __launch_bounds__(256) void count_kernel(const int* __restrict__ rows,
                                                    int* __restrict__ row_ptr) {
    for (int e = blockIdx.x * blockDim.x + threadIdx.x; e < N_EDGES;
         e += gridDim.x * blockDim.x)
        atomicAdd(&row_ptr[rows[e] + 1], 1);
}

// ---------------------- single-block inclusive scan over row_ptr[1..N_NODES]
__global__ __launch_bounds__(1024) void scan_kernel(int* __restrict__ row_ptr) {
    __shared__ int buf[1024];
    __shared__ int carry;
    if (threadIdx.x == 0) carry = 0;
    __syncthreads();
    for (int base = 1; base <= N_NODES; base += 1024) {
        int i = base + (int)threadIdx.x;
        int v = (i <= N_NODES) ? row_ptr[i] : 0;
        buf[threadIdx.x] = v;
        __syncthreads();
        for (int off = 1; off < 1024; off <<= 1) {
            int t = (threadIdx.x >= (unsigned)off) ? buf[threadIdx.x - off] : 0;
            __syncthreads();
            buf[threadIdx.x] += t;
            __syncthreads();
        }
        int inc = buf[threadIdx.x] + carry;
        if (i <= N_NODES) row_ptr[i] = inc;
        __syncthreads();
        if (threadIdx.x == 1023) carry = inc;
        __syncthreads();
    }
}

__global__ void copy_int(const int* __restrict__ src, int* __restrict__ dst, int n) {
    for (int i = blockIdx.x * blockDim.x + threadIdx.x; i < n;
         i += gridDim.x * blockDim.x) dst[i] = src[i];
}

// --------------------------------------- CSR build: scatter edge ids to rows
__global__ __launch_bounds__(256) void scatter_kernel(const int* __restrict__ rows,
                                                      int* __restrict__ cursor,
                                                      int* __restrict__ perm) {
    for (int e = blockIdx.x * blockDim.x + threadIdx.x; e < N_EDGES;
         e += gridDim.x * blockDim.x) {
        int pos = atomicAdd(&cursor[rows[e]], 1);
        perm[pos] = e;
    }
}

// ----------------- S[r] = sum_{e in row r} vals[e] * X[cols[e]]  (no atomics)
__global__ __launch_bounds__(128) void spmm_gather(
        const float* __restrict__ X, const int* __restrict__ cols,
        const float* __restrict__ vals, const int* __restrict__ row_ptr,
        const int* __restrict__ perm, float* __restrict__ S) {
    const int r = blockIdx.x;
    const int t = threadIdx.x;           // 128 threads x float4 = 512 dims
    const int beg = row_ptr[r];
    const int end = row_ptr[r + 1];
    v4f acc = {};
    for (int i = beg; i < end; ++i) {
        const int e = perm[i];
        if (i + 1 < end) {               // prefetch next source row
            const int en = perm[i + 1];
            __builtin_prefetch(X + (size_t)cols[en] * IN_DIM + t * 4, 0, 3);
        }
        const float v = vals[e];
        const v4f x = *(const v4f*)(X + (size_t)cols[e] * IN_DIM + t * 4);
#pragma unroll
        for (int j = 0; j < 4; ++j) acc[j] += v * x[j];
    }
    *(v4f*)(S + (size_t)r * IN_DIM + t * 4) = acc;
}

// --------------------- h1 = relu(S @ W1 + b1), in place over S (bf16 WMMA)
// block = 8 waves: 2 row-groups x 4 col-groups; tile = 32 rows x 512 cols.
__global__ __launch_bounds__(256) void gemm_bias_relu(
        float* __restrict__ S, const __bf16* __restrict__ W1t,
        const float* __restrict__ b1) {
    const int lane = threadIdx.x & 31;
    const int wave = threadIdx.x >> 5;   // 0..7
    const int wrow = wave >> 2;          // 0..1
    const int wcol = wave & 3;           // 0..3
    const int m15  = lane & 15;
    const int half = lane >> 4;          // 0 or 1

    const int rowBase = blockIdx.x * 32 + wrow * 16;
    int arow = rowBase + m15;
    if (arow >= N_NODES) arow = N_NODES - 1;      // clamp: keeps EXEC full
    const int colBase = wcol * 128;

    v8f zero = {};
    v8f acc[8];
#pragma unroll
    for (int j = 0; j < 8; ++j) acc[j] = zero;

    const float* Srow = S + (size_t)arow * IN_DIM;

    for (int k0 = 0; k0 < IN_DIM; k0 += 32) {
        // A fragment: elems 0..15 -> K=k0+half*16+0..15
        v4f a0 = *(const v4f*)(Srow + k0 + half * 16);
        v4f a1 = *(const v4f*)(Srow + k0 + half * 16 + 4);
        v4f a2 = *(const v4f*)(Srow + k0 + half * 16 + 8);
        v4f a3 = *(const v4f*)(Srow + k0 + half * 16 + 12);
        v16bf a;
#pragma unroll
        for (int i = 0; i < 4; ++i) {
            a[i]      = (__bf16)a0[i];
            a[4 + i]  = (__bf16)a1[i];
            a[8 + i]  = (__bf16)a2[i];
            a[12 + i] = (__bf16)a3[i];
        }
        // B fragments: lanes 0-15 K=0..15, lanes 16-31 K=16..31 (W1t row-contig)
#pragma unroll
        for (int j = 0; j < 8; ++j) {
            const int ncol = colBase + j * 16 + m15;
            v16bf b = *(const v16bf*)(W1t + (size_t)ncol * IN_DIM + k0 + half * 16);
            acc[j] = __builtin_amdgcn_wmma_f32_16x16x32_bf16(
                false, a, false, b, (short)0, acc[j], false, false);
        }
    }

    __syncthreads();   // all waves done reading S rows before in-place store

    // C layout: VGPR v, lane L -> m = v + 8*(L>>4), n = L&15
#pragma unroll
    for (int j = 0; j < 8; ++j) {
        const int n = colBase + j * 16 + m15;
        const float bias = b1[n];
#pragma unroll
        for (int v = 0; v < 8; ++v) {
            const int m = rowBase + v + half * 8;
            if (m < N_NODES) {
                float val = acc[j][v] + bias;
                S[(size_t)m * IN_DIM + n] = val > 0.0f ? val : 0.0f;
            }
        }
    }
}

// ------- u[b] = sum over CSR row lang_idx[b] of vals[e] * h1[cols[e]]
__global__ __launch_bounds__(128) void edge_select_csr(
        const int* __restrict__ cols, const float* __restrict__ vals,
        const int* __restrict__ row_ptr, const int* __restrict__ perm,
        const int* __restrict__ lang_idx, const float* __restrict__ h1,
        float* __restrict__ u) {
    const int b = blockIdx.x;            // 0..BATCH-1
    const int t = threadIdx.x;           // 128 threads x float4
    const int node = lang_idx[b];
    const int beg = row_ptr[node];
    const int end = row_ptr[node + 1];
    v4f acc = {};
    for (int i = beg; i < end; ++i) {
        const int e = perm[i];
        if (i + 1 < end) {
            const int en = perm[i + 1];
            __builtin_prefetch(h1 + (size_t)cols[en] * IN_DIM + t * 4, 0, 3);
        }
        const float v = vals[e];
        const v4f x = *(const v4f*)(h1 + (size_t)cols[e] * IN_DIM + t * 4);
#pragma unroll
        for (int j = 0; j < 4; ++j) acc[j] += v * x[j];
    }
    *(v4f*)(u + b * HID + t * 4) = acc;
}

// ------------------------- h2 = relu(u @ W2 + b2)   (16 x 512, tiny)
__global__ void mlp2_kernel(const float* __restrict__ u,
                            const float* __restrict__ W2,
                            const float* __restrict__ b2,
                            float* __restrict__ h2) {
    int idx = blockIdx.x * blockDim.x + threadIdx.x;  // BATCH*HID
    int b = idx >> 9;
    int j = idx & 511;
    float acc = b2[j];
    const float* ub = u + b * HID;
    for (int k = 0; k < HID; ++k) acc += ub[k] * W2[k * HID + j];
    h2[idx] = acc > 0.0f ? acc : 0.0f;
}

// ------------------------- out = h2 @ Wf + bf        (16 x 256, tiny)
__global__ void out_kernel(const float* __restrict__ h2,
                           const float* __restrict__ Wf,
                           const float* __restrict__ bfv,
                           float* __restrict__ out) {
    int idx = blockIdx.x * blockDim.x + threadIdx.x;  // BATCH*OUT_DIM
    int b = idx >> 8;
    int o = idx & 255;
    float acc = bfv[o];
    const float* hb = h2 + b * HID;
    for (int j = 0; j < HID; ++j) acc += hb[j] * Wf[j * OUT_DIM + o];
    out[idx] = acc;
}

extern "C" void kernel_launch(void* const* d_in, const int* in_sizes, int n_in,
                              void* d_out, int out_size, void* d_ws, size_t ws_size,
                              hipStream_t stream) {
    const float* X    = (const float*)d_in[0];
    const int*   rows = (const int*)  d_in[1];
    const int*   cols = (const int*)  d_in[2];
    const float* vals = (const float*)d_in[3];
    const float* W1   = (const float*)d_in[4];
    const float* b1   = (const float*)d_in[5];
    const float* W2   = (const float*)d_in[6];
    const float* b2   = (const float*)d_in[7];
    const float* Wf   = (const float*)d_in[8];
    const float* bfv  = (const float*)d_in[9];
    const int*   lang = (const int*)  d_in[10];

    char* ws = (char*)d_ws;
    const size_t SZ_S   = (size_t)N_NODES * IN_DIM * sizeof(float);  // 204.8 MB
    const size_t SZ_W1T = (size_t)IN_DIM * HID * 2;                  // 512 KB
    const size_t SZ_RP  = 400128;                                    // (N+1)*4 padded
    const size_t SZ_CUR = 400128;                                    // N*4 padded
    const size_t SZ_PRM = (size_t)N_EDGES * 4;                       // 13.2 MB

    float*  S       = (float*) ws;                                   // S, later h1
    __bf16* W1t     = (__bf16*)(ws + SZ_S);
    int*    row_ptr = (int*)   (ws + SZ_S + SZ_W1T);
    int*    cursor  = (int*)   (ws + SZ_S + SZ_W1T + SZ_RP);
    int*    perm    = (int*)   (ws + SZ_S + SZ_W1T + SZ_RP + SZ_CUR);
    float*  u       = (float*) (ws + SZ_S + SZ_W1T + SZ_RP + SZ_CUR + SZ_PRM);
    float*  h2      = u + BATCH * HID;
    float*  out     = (float*)d_out;

    // CSR build
    zero_int<<<512, 256, 0, stream>>>(row_ptr, N_NODES + 1);
    conv_w1t<<<(IN_DIM * HID) / 256, 256, 0, stream>>>(W1, W1t);
    count_kernel<<<4096, 256, 0, stream>>>(rows, row_ptr);
    scan_kernel<<<1, 1024, 0, stream>>>(row_ptr);
    copy_int<<<512, 256, 0, stream>>>(row_ptr, cursor, N_NODES);
    scatter_kernel<<<4096, 256, 0, stream>>>(rows, cursor, perm);

    // layer 1: spmm-gather then fused WMMA GEMM + bias + relu (in place)
    spmm_gather<<<N_NODES, 128, 0, stream>>>(X, cols, vals, row_ptr, perm, S);
    gemm_bias_relu<<<(N_NODES + 31) / 32, 256, 0, stream>>>(S, W1t, b1);

    // layer 2 collapsed to the 16 requested rows (spmm/linear commute)
    edge_select_csr<<<BATCH, 128, 0, stream>>>(cols, vals, row_ptr, perm, lang, S, u);
    mlp2_kernel<<<(BATCH * HID) / 128, 128, 0, stream>>>(u, W2, b2, h2);
    out_kernel<<<(BATCH * OUT_DIM) / 128, 128, 0, stream>>>(h2, Wf, bfv, out);
}
